// Cross_Attention_Add_19507741458638
// MI455X (gfx1250) — matrix operand order repaired and checked by reading, not compiled, not run on hardware
//
#include <hip/hip_runtime.h>
#include <hip/hip_bf16.h>

// ---------------------------------------------------------------------------
// CDNA5 (gfx1250) wave32 WMMA GEMM building blocks (bf16 A/B, f32 accum)
// ---------------------------------------------------------------------------
typedef __attribute__((ext_vector_type(16))) __bf16 v16bf;
typedef __attribute__((ext_vector_type(8)))  __bf16 v8bf;
typedef __attribute__((ext_vector_type(8)))  float  v8f;

#define LDA 136   // padded LDS stride (elems) for K=128 tiles: 272B -> bank-spread
#define LDU 264   // padded LDS stride for K=256 tiles

__device__ __forceinline__ __bf16 f2bf(float f) {
    union { float f; unsigned u; } v; v.f = f;
    unsigned r = v.u + 0x7FFFu + ((v.u >> 16) & 1u);   // RNE
    unsigned short h = (unsigned short)(r >> 16);
    __bf16 out; __builtin_memcpy(&out, &h, 2);
    return out;
}

// One 16x16 output tile accumulated over K (multiple of 32).
// A: 16 x K row-major bf16 in LDS (stride lda).
// Bt: N x K bf16 in LDS, weight stored TRANSPOSED (row n = output column n).
// Fragment layouts per CDNA5 ISA 7.12.2:
//   A (16-bit 16x32): lanes 0-15 hold M=lane, K = {0..7}U{16..23};
//                     lanes 16-31 hold M=lane-16, K = {8..15}U{24..31}
//   B: lane n<16 -> col n, K = {0..7}U{16..23} ; lane n>=16 -> col n-16, K = {8..15}U{24..31}
__device__ __forceinline__ v8f wmma_tile(const __bf16* A, int lda,
                                         const __bf16* Bt, int ldb,
                                         int K, int n0, v8f acc) {
    const int lane = threadIdx.x & 31;
    const int half = lane >> 4;
    const int mn   = lane & 15;
    const __bf16* arow = A  + mn * lda + (half ? 8 : 0);
    const __bf16* brow = Bt + (n0 + mn) * ldb + (half ? 8 : 0);
#pragma unroll
    for (int ks = 0; ks < K; ks += 32) {
        v8bf a0 = *(const v8bf*)(arow + ks);
        v8bf a1 = *(const v8bf*)(arow + ks + 16);
        v8bf b0 = *(const v8bf*)(brow + ks);
        v8bf b1 = *(const v8bf*)(brow + ks + 16);
        v16bf av = __builtin_shufflevector(a0, a1, 0,1,2,3,4,5,6,7,8,9,10,11,12,13,14,15);
        v16bf bv = __builtin_shufflevector(b0, b1, 0,1,2,3,4,5,6,7,8,9,10,11,12,13,14,15);
        acc = __builtin_amdgcn_wmma_f32_16x16x32_bf16(false, av, false, bv,
                                                      (short)0, acc, false, false);
    }
    return acc;
}

// ---------------------------------------------------------------------------
// Kernel 0: zero aggr buffer
// ---------------------------------------------------------------------------
__global__ __launch_bounds__(256) void zero_kernel(float4* p, long long n4) {
    long long i = (long long)blockIdx.x * blockDim.x + threadIdx.x;
    if (i < n4) p[i] = make_float4(0.f, 0.f, 0.f, 0.f);
}

// ---------------------------------------------------------------------------
// Kernel 1: Q/K/V GEMMs fused -> h = relu((Q+bq)*(K+bk)), V  (64 rows / block)
// ---------------------------------------------------------------------------
__global__ __launch_bounds__(256) void qkv_h_kernel(
    const float* __restrict__ x, const float* __restrict__ prompt,
    const float* __restrict__ Wq, const float* __restrict__ bq,
    const float* __restrict__ Wk, const float* __restrict__ bk,
    const float* __restrict__ Wv, const float* __restrict__ bv,
    float* __restrict__ h_ws, float* __restrict__ v_ws, int N)
{
    extern __shared__ char smem[];
    __bf16* wq_t = (__bf16*)smem;              // 128*LDA
    __bf16* wk_t = wq_t + 128 * LDA;
    __bf16* wv_t = wk_t + 128 * LDA;
    __bf16* ax   = wv_t + 128 * LDA;           // 64*LDA
    __bf16* apm  = ax   + 64 * LDA;            // 64*LDA

    // stage weights transposed (coalesced global reads)
    for (int idx = threadIdx.x; idx < 128 * 128; idx += 256) {
        int k = idx >> 7, n = idx & 127;
        wq_t[n * LDA + k] = f2bf(Wq[idx]);
        wk_t[n * LDA + k] = f2bf(Wk[idx]);
        wv_t[n * LDA + k] = f2bf(Wv[idx]);
    }
    const int row0 = blockIdx.x * 64;
    for (int idx = threadIdx.x; idx < 64 * 128; idx += 256) {
        int r = idx >> 7, c = idx & 127;
        int row = row0 + r;
        float xv = 0.f, pv = 0.f;
        if (row < N) { xv = x[(size_t)row * 128 + c]; pv = prompt[(size_t)row * 128 + c]; }
        ax [r * LDA + c] = f2bf(xv);
        apm[r * LDA + c] = f2bf(pv);
    }
    __syncthreads();

    const int wave = threadIdx.x >> 5, lane = threadIdx.x & 31;
    const int n0 = wave * 16, half = lane >> 4;
    const int col = n0 + (lane & 15);
    const float bqv = bq[col], bkv = bk[col], bvv = bv[col];

    for (int sub = 0; sub < 4; ++sub) {
        const __bf16* Ax = ax  + sub * 16 * LDA;
        const __bf16* Ap = apm + sub * 16 * LDA;
        v8f z = {};
        v8f q = wmma_tile(Ax, LDA, wq_t, LDA, 128, n0, z);
        v8f k = wmma_tile(Ap, LDA, wk_t, LDA, 128, n0, z);
        v8f v = wmma_tile(Ap, LDA, wv_t, LDA, 128, n0, z);
#pragma unroll
        for (int r = 0; r < 8; ++r) {
            int row = row0 + sub * 16 + half * 8 + r;
            if (row < N) {
                float hv = fmaxf((q[r] + bqv) * (k[r] + bkv), 0.f);
                h_ws[(size_t)row * 128 + col] = hv;
                v_ws[(size_t)row * 128 + col] = v[r] + bvv;
            }
        }
    }
}

// ---------------------------------------------------------------------------
// Kernel 2: edge MLP (2 GEMMs) + gather h[src] + msg + atomic scatter-add
// ---------------------------------------------------------------------------
__global__ __launch_bounds__(256) void edge_kernel(
    const float* __restrict__ edge_attr, const float* __restrict__ prompt_e,
    const int* __restrict__ edge_index,
    const float* __restrict__ We1, const float* __restrict__ be1,
    const float* __restrict__ We2, const float* __restrict__ be2,
    const float* __restrict__ h_ws, float* __restrict__ aggr, int E)
{
    extern __shared__ char smem[];
    __bf16* we1_t = (__bf16*)smem;             // 128*LDA
    __bf16* we2_t = we1_t + 128 * LDA;
    __bf16* a1    = we2_t + 128 * LDA;         // 64*LDA
    __bf16* a2    = a1    + 64 * LDA;
    __bf16* pe    = a2    + 64 * LDA;
    int* sidx = (int*)(pe + 64 * LDA);         // 64
    int* didx = sidx + 64;                     // 64

    for (int idx = threadIdx.x; idx < 128 * 128; idx += 256) {
        int k = idx >> 7, n = idx & 127;
        we1_t[n * LDA + k] = f2bf(We1[idx]);
        we2_t[n * LDA + k] = f2bf(We2[idx]);
    }
    const int e0 = blockIdx.x * 64;
    for (int idx = threadIdx.x; idx < 64 * 128; idx += 256) {
        int r = idx >> 7, c = idx & 127;
        int e = e0 + r;
        float ea = 0.f, pv = 0.f;
        if (e < E) { ea = edge_attr[(size_t)e * 128 + c]; pv = prompt_e[(size_t)e * 128 + c]; }
        pe[r * LDA + c] = f2bf(pv);
        a1[r * LDA + c] = f2bf(ea + pv);
    }
    if (threadIdx.x < 64) {
        int e = e0 + threadIdx.x;
        sidx[threadIdx.x] = (e < E) ? edge_index[e] : 0;
        didx[threadIdx.x] = (e < E) ? edge_index[E + e] : 0;
    }
    __syncthreads();

    const int wave = threadIdx.x >> 5, lane = threadIdx.x & 31;
    const int n0 = wave * 16, half = lane >> 4;
    const int col = n0 + (lane & 15);
    const float b1v = be1[col], b2v = be2[col];

    // GEMM 1: e1 = relu((edge_attr+pe)@We1 + be1); a2 = bf16(e1 + pe)
    for (int sub = 0; sub < 4; ++sub) {
        v8f z = {};
        v8f acc = wmma_tile(a1 + sub * 16 * LDA, LDA, we1_t, LDA, 128, n0, z);
#pragma unroll
        for (int r = 0; r < 8; ++r) {
            int rr = sub * 16 + half * 8 + r;
            float ev = fmaxf(acc[r] + b1v, 0.f) + (float)pe[rr * LDA + col];
            a2[rr * LDA + col] = f2bf(ev);
        }
    }
    __syncthreads();

    // GEMM 2 + gather + msg + scatter-add
    for (int sub = 0; sub < 4; ++sub) {
        v8f z = {};
        v8f acc = wmma_tile(a2 + sub * 16 * LDA, LDA, we2_t, LDA, 128, n0, z);
#pragma unroll
        for (int r = 0; r < 8; ++r) {
            int m = sub * 16 + half * 8 + r;
            int e = e0 + m;
            if (e < E) {
                float e2 = fmaxf(acc[r] + b2v, 0.f);
                float srcv = h_ws[(size_t)sidx[m] * 128 + col];   // L2-resident gather
                float msg = fmaxf(srcv + e2, 0.f);
                unsafeAtomicAdd(&aggr[(size_t)didx[m] * 128 + col], msg);
            }
        }
    }
}

// ---------------------------------------------------------------------------
// Kernel 3: t = relu((h+V+aggr)@Wh + bh) + x   (t written in-place over h)
// ---------------------------------------------------------------------------
__global__ __launch_bounds__(256) void node_out_kernel(
    const float* __restrict__ x,
    const float* h_ws, const float* __restrict__ v_ws,
    const float* __restrict__ aggr,
    const float* __restrict__ Wh, const float* __restrict__ bh,
    float* t_ws, int N)
{
    extern __shared__ char smem[];
    __bf16* wh_t = (__bf16*)smem;              // 128*LDA
    __bf16* a    = wh_t + 128 * LDA;           // 64*LDA

    for (int idx = threadIdx.x; idx < 128 * 128; idx += 256) {
        int k = idx >> 7, n = idx & 127;
        wh_t[n * LDA + k] = f2bf(Wh[idx]);
    }
    const int row0 = blockIdx.x * 64;
    for (int idx = threadIdx.x; idx < 64 * 128; idx += 256) {
        int r = idx >> 7, c = idx & 127;
        int row = row0 + r;
        float v = 0.f;
        if (row < N) {
            size_t o = (size_t)row * 128 + c;
            v = h_ws[o] + v_ws[o] + aggr[o];
        }
        a[r * LDA + c] = f2bf(v);
    }
    __syncthreads();

    const int wave = threadIdx.x >> 5, lane = threadIdx.x & 31;
    const int n0 = wave * 16, half = lane >> 4;
    const int col = n0 + (lane & 15);
    const float bhv = bh[col];

    for (int sub = 0; sub < 4; ++sub) {
        v8f z = {};
        v8f acc = wmma_tile(a + sub * 16 * LDA, LDA, wh_t, LDA, 128, n0, z);
#pragma unroll
        for (int r = 0; r < 8; ++r) {
            int row = row0 + sub * 16 + half * 8 + r;
            if (row < N) {
                size_t o = (size_t)row * 128 + col;
                t_ws[o] = fmaxf(acc[r] + bhv, 0.f) + x[o];
            }
        }
    }
}

// ---------------------------------------------------------------------------
// Kernel 4: FFN: out = relu(t@Wf1+bf1) @ Wf2 + bf2   (128 -> 256 -> 128)
// ---------------------------------------------------------------------------
__global__ __launch_bounds__(256) void ffn_kernel(
    const float* __restrict__ t_ws,
    const float* __restrict__ Wf1, const float* __restrict__ bf1,
    const float* __restrict__ Wf2, const float* __restrict__ bf2,
    float* __restrict__ out, int N)
{
    extern __shared__ char smem[];
    __bf16* wf1_t = (__bf16*)smem;             // 256 cols x K=128 : 256*LDA
    __bf16* wf2_t = wf1_t + 256 * LDA;         // 128 cols x K=256 : 128*LDU
    __bf16* u     = wf2_t + 128 * LDU;         // 64 x 256 : 64*LDU
    __bf16* a     = u     + 64  * LDU;         // 64*LDA

    for (int idx = threadIdx.x; idx < 128 * 256; idx += 256) {
        int k = idx >> 8, n = idx & 255;       // Wf1[k][n], k<128, n<256
        wf1_t[n * LDA + k] = f2bf(Wf1[idx]);
    }
    for (int idx = threadIdx.x; idx < 256 * 128; idx += 256) {
        int k = idx >> 7, n = idx & 127;       // Wf2[k][n], k<256, n<128
        wf2_t[n * LDU + k] = f2bf(Wf2[idx]);
    }
    const int row0 = blockIdx.x * 64;
    for (int idx = threadIdx.x; idx < 64 * 128; idx += 256) {
        int r = idx >> 7, c = idx & 127;
        int row = row0 + r;
        a[r * LDA + c] = f2bf((row < N) ? t_ws[(size_t)row * 128 + c] : 0.f);
    }
    __syncthreads();

    const int wave = threadIdx.x >> 5, lane = threadIdx.x & 31;
    const int half = lane >> 4;

    // stage 1: u = relu(a @ Wf1 + bf1)  (each wave covers cols w*16 and w*16+128)
    for (int sub = 0; sub < 4; ++sub) {
#pragma unroll
        for (int part = 0; part < 2; ++part) {
            int n0 = wave * 16 + part * 128;
            int col = n0 + (lane & 15);
            v8f z = {};
            v8f acc = wmma_tile(a + sub * 16 * LDA, LDA, wf1_t, LDA, 128, n0, z);
            float b1 = bf1[col];
#pragma unroll
            for (int r = 0; r < 8; ++r) {
                int rr = sub * 16 + half * 8 + r;
                u[rr * LDU + col] = f2bf(fmaxf(acc[r] + b1, 0.f));
            }
        }
    }
    __syncthreads();

    // stage 2: out = u @ Wf2 + bf2   (K = 256)
    const int n0 = wave * 16;
    const int col = n0 + (lane & 15);
    const float b2 = bf2[col];
    for (int sub = 0; sub < 4; ++sub) {
        v8f z = {};
        v8f acc = wmma_tile(u + sub * 16 * LDU, LDU, wf2_t, LDU, 256, n0, z);
#pragma unroll
        for (int r = 0; r < 8; ++r) {
            int row = row0 + sub * 16 + half * 8 + r;
            if (row < N)
                out[(size_t)row * 128 + col] = acc[r] + b2;
        }
    }
}

// ---------------------------------------------------------------------------
// Host launcher
// ---------------------------------------------------------------------------
extern "C" void kernel_launch(void* const* d_in, const int* in_sizes, int n_in,
                              void* d_out, int out_size, void* d_ws, size_t ws_size,
                              hipStream_t stream) {
    const float* x         = (const float*)d_in[0];
    const float* prompt    = (const float*)d_in[1];
    const float* prompt_e  = (const float*)d_in[2];
    const float* edge_attr = (const float*)d_in[3];
    const int*   edge_idx  = (const int*)  d_in[4];
    const float* Wq  = (const float*)d_in[5];  const float* bq  = (const float*)d_in[6];
    const float* Wk  = (const float*)d_in[7];  const float* bk  = (const float*)d_in[8];
    const float* Wv  = (const float*)d_in[9];  const float* bv  = (const float*)d_in[10];
    const float* We1 = (const float*)d_in[11]; const float* be1 = (const float*)d_in[12];
    const float* We2 = (const float*)d_in[13]; const float* be2 = (const float*)d_in[14];
    const float* Wh  = (const float*)d_in[15]; const float* bh  = (const float*)d_in[16];
    const float* Wf1 = (const float*)d_in[17]; const float* bf1 = (const float*)d_in[18];
    const float* Wf2 = (const float*)d_in[19]; const float* bf2 = (const float*)d_in[20];

    const int N = in_sizes[0] / 128;
    const int E = in_sizes[3] / 128;
    const size_t nd = (size_t)N * 128;

    float* h_ws  = (float*)d_ws;          // [N,128]; later reused as t
    float* v_ws  = h_ws + nd;             // [N,128]
    float* aggr  = v_ws + nd;             // [N,128]
    float* outp  = (float*)d_out;

    const int nodeBlocks = (N + 63) / 64;
    const int edgeBlocks = (E + 63) / 64;

    // LDS byte sizes
    const size_t SM_A = (size_t)(3 * 128 * LDA + 2 * 64 * LDA) * 2;                  // 139264
    const size_t SM_B = (size_t)(2 * 128 * LDA + 3 * 64 * LDA) * 2 + 128 * 4;        // 122368
    const size_t SM_C = (size_t)(128 * LDA + 64 * LDA) * 2;                          //  52224
    const size_t SM_D = (size_t)(256 * LDA + 128 * LDU + 64 * LDU + 64 * LDA) * 2;   // 188416

    long long n4 = (long long)nd / 4;
    zero_kernel<<<(int)((n4 + 255) / 256), 256, 0, stream>>>((float4*)aggr, n4);

    qkv_h_kernel<<<nodeBlocks, 256, SM_A, stream>>>(x, prompt, Wq, bq, Wk, bk, Wv, bv,
                                                    h_ws, v_ws, N);

    edge_kernel<<<edgeBlocks, 256, SM_B, stream>>>(edge_attr, prompt_e, edge_idx,
                                                   We1, be1, We2, be2, h_ws, aggr, E);

    node_out_kernel<<<nodeBlocks, 256, SM_C, stream>>>(x, h_ws, v_ws, aggr, Wh, bh,
                                                       /*t in-place*/ h_ws, N);

    ffn_kernel<<<nodeBlocks, 256, SM_D, stream>>>(h_ws, Wf1, bf1, Wf2, bf2, outp, N);
}
